// LSTM_39737037422825
// MI455X (gfx1250) — matrix operand order repaired and checked by reading, not compiled, not run on hardware
//
#include <hip/hip_runtime.h>
#include <math.h>

// ---------------------------------------------------------------------------
// 2-layer LSTM for MI455X (gfx1250, wave32, WMMA bf16 16x16x32)
//   phase A (per layer): gx = X * W_ih^T + b     -- big parallel WMMA GEMM
//   phase B (per layer): persistent scan kernel, one grid barrier per step,
//                        c kept in registers, h streamed through L2 as bf16
// ---------------------------------------------------------------------------

typedef __bf16 bf16_t;
typedef __attribute__((ext_vector_type(16))) __bf16 v16bf;
typedef __attribute__((ext_vector_type(8)))  __bf16 bf16x8;
typedef __attribute__((ext_vector_type(8)))  float  v8f;
typedef __attribute__((ext_vector_type(4)))  int    i32x4;

#define B_   32
#define T_   512
#define D_   1024      // == H, both GEMM K dims are 1024
#define H_   1024
#define G4H  4096
#define NWG_SCAN 64

union FragAB {
    v16bf  v;
    bf16x8 h[2];
    i32x4  q[2];
};

// --------------------------- helpers ---------------------------------------

__global__ void init_ws_kernel(unsigned* counters) {
    if (threadIdx.x < 2) counters[threadIdx.x] = 0u;
}

__global__ void cvt_f32_to_bf16_kernel(const float* __restrict__ src,
                                       bf16_t* __restrict__ dst, int n) {
    int i = blockIdx.x * blockDim.x + threadIdx.x;
    if (i < n) dst[i] = (bf16_t)src[i];
}

// --------------------------- phase A: input GEMM ----------------------------
// gx[m, n] = sum_k A[srcRow(m), k] * W[n, k] + bias[n]
//   m = t*B + b ; srcRow = t*strideT + b*strideB (handles batch-first x)
//   A: bf16 row-major [.., K=1024]; W: bf16 row-major [4096, 1024]
// Each wave: 32x32 output tile (2x2 fragments), block = 4 waves = 64x64 tile.
__global__ void __launch_bounds__(128)
gemm_gx_kernel(const bf16_t* __restrict__ A, const bf16_t* __restrict__ W,
               const float* __restrict__ bias, float* __restrict__ gx,
               int strideT, int strideB) {
    const int lane  = threadIdx.x & 31;
    const int wave  = threadIdx.x >> 5;
    const int mTile = blockIdx.x * 64 + (wave & 1) * 32;
    const int nTile = blockIdx.y * 64 + (wave >> 1) * 32;
    const int half  = lane >> 4;     // K-half select (A/B fragment layout)
    const int l15   = lane & 15;     // A: M row ; B: N column ; C/D: N column

    // source rows for the two M fragments (undo batch-first layout)
    int srcRow[2];
#pragma unroll
    for (int mf = 0; mf < 2; ++mf) {
        int m = mTile + mf * 16 + l15;
        int t = m / B_;
        int b = m - t * B_;
        srcRow[mf] = t * strideT + b * strideB;
    }

    v8f acc[2][2] = {};

    for (int k0 = 0; k0 < D_; k0 += 32) {
        FragAB a[2], bw[2];
#pragma unroll
        for (int mf = 0; mf < 2; ++mf) {
            const bf16_t* p = A + (long)srcRow[mf] * D_ + k0 + half * 8;
            a[mf].q[0] = *(const i32x4*)(p);        // K chunk 0..7 (this half)
            a[mf].q[1] = *(const i32x4*)(p + 16);   // K chunk 16..23 (this half)
        }
#pragma unroll
        for (int nf = 0; nf < 2; ++nf) {
            const bf16_t* p = W + (long)(nTile + nf * 16 + l15) * D_ + k0 + half * 8;
            bw[nf].q[0] = *(const i32x4*)(p);       // K chunk 0..7 (this half)
            bw[nf].q[1] = *(const i32x4*)(p + 16);
        }
#pragma unroll
        for (int mf = 0; mf < 2; ++mf)
#pragma unroll
            for (int nf = 0; nf < 2; ++nf)
                acc[mf][nf] = __builtin_amdgcn_wmma_f32_16x16x32_bf16(
                    false, a[mf].v, false, bw[nf].v, (short)0, acc[mf][nf],
                    false, false);
    }

    // epilogue: C/D layout m = base + half*8 + r, n = base + l15
#pragma unroll
    for (int mf = 0; mf < 2; ++mf)
#pragma unroll
        for (int nf = 0; nf < 2; ++nf) {
            int n = nTile + nf * 16 + l15;
            float bv = bias[n];
#pragma unroll
            for (int r = 0; r < 8; ++r) {
                int m = mTile + mf * 16 + half * 8 + r;
                gx[(long)m * G4H + n] = acc[mf][nf][r] + bv;
            }
        }
}

// --------------------------- phase B: persistent scan -----------------------
// Grid = 64 blocks x 128 threads. Block owns 16 H-columns; wave w computes
// gate w's 32x16 tile of (h_{t-1} @ W_hh^T) via WMMA, tiles meet in LDS,
// elementwise LSTM cell update keeps c in registers (4 floats/thread).
// One monotonic atomic grid barrier per step.
__global__ void __launch_bounds__(128)
lstm_scan_kernel(const float* __restrict__ gx, const bf16_t* __restrict__ Whh,
                 bf16_t* __restrict__ hs, float* __restrict__ out,
                 float* __restrict__ hn, float* __restrict__ cn,
                 unsigned* __restrict__ counter, int writeOut) {
    __shared__ float sg[4][B_][16];

    const int tid   = threadIdx.x;
    const int lane  = tid & 31;
    const int gate  = tid >> 5;              // wave id == gate (i,f,g,o)
    const int nTile = blockIdx.x * 16;
    const int half  = lane >> 4;
    const int l15   = lane & 15;
    const long wrow = (long)(gate * H_ + nTile + l15) * H_;  // W_hh row base

    float creg[4] = {0.0f, 0.0f, 0.0f, 0.0f};
    unsigned gen = 0;

    for (int t = 0; t < T_; ++t) {
        if (t + 1 < T_)   // warm next step's gate pre-activations into cache
            __builtin_prefetch(gx + (long)(t + 1) * (B_ * G4H) + tid * 128, 0, 1);

        v8f acc0 = {}, acc1 = {};
        if (t > 0) {
            const bf16_t* hprev = hs + (long)(t - 1) * (B_ * H_);
            for (int k0 = 0; k0 < H_; k0 += 32) {
                FragAB a0, a1, bw;
                const bf16_t* pa0 = hprev + (long)l15 * H_ + k0 + half * 8;
                a0.q[0] = *(const i32x4*)(pa0);
                a0.q[1] = *(const i32x4*)(pa0 + 16);
                const bf16_t* pa1 = hprev + (long)(16 + l15) * H_ + k0 + half * 8;
                a1.q[0] = *(const i32x4*)(pa1);
                a1.q[1] = *(const i32x4*)(pa1 + 16);
                const bf16_t* pb = Whh + wrow + k0 + half * 8;
                bw.q[0] = *(const i32x4*)(pb);
                bw.q[1] = *(const i32x4*)(pb + 16);
                acc0 = __builtin_amdgcn_wmma_f32_16x16x32_bf16(
                    false, a0.v, false, bw.v, (short)0, acc0, false, false);
                acc1 = __builtin_amdgcn_wmma_f32_16x16x32_bf16(
                    false, a1.v, false, bw.v, (short)0, acc1, false, false);
            }
        }

        // gate tile -> LDS (C/D layout: m = half*8 + r, n = l15)
#pragma unroll
        for (int r = 0; r < 8; ++r) {
            sg[gate][half * 8 + r][l15]      = acc0[r];
            sg[gate][16 + half * 8 + r][l15] = acc1[r];
        }
        __syncthreads();

        // elementwise LSTM cell: 512 elements / 128 threads = 4 each
        const float* gxt = gx + (long)t * (B_ * G4H);
#pragma unroll
        for (int j = 0; j < 4; ++j) {
            int e = tid * 4 + j;
            int b = e >> 4;
            int n = e & 15;
            int col = nTile + n;
            float iv = sg[0][b][n] + gxt[b * G4H + col];
            float fv = sg[1][b][n] + gxt[b * G4H + H_ + col];
            float gv = sg[2][b][n] + gxt[b * G4H + 2 * H_ + col];
            float ov = sg[3][b][n] + gxt[b * G4H + 3 * H_ + col];
            iv = 1.0f / (1.0f + __expf(-iv));
            fv = 1.0f / (1.0f + __expf(-fv));
            gv = tanhf(gv);
            ov = 1.0f / (1.0f + __expf(-ov));
            float cnew = fv * creg[j] + iv * gv;
            creg[j] = cnew;
            float hnew = ov * tanhf(cnew);
            hs[(long)t * (B_ * H_) + b * H_ + col] = (bf16_t)hnew;   // next step A / next layer X
            if (writeOut) out[(long)b * (T_ * H_) + t * H_ + col] = hnew;
            if (t == T_ - 1) hn[b * H_ + col] = hnew;
        }
        __syncthreads();

        // grid-wide barrier: all h_t writes visible before any block reads them
        if (tid == 0) {
            __threadfence();
            __hip_atomic_fetch_add(counter, 1u, __ATOMIC_ACQ_REL,
                                   __HIP_MEMORY_SCOPE_AGENT);
            gen += NWG_SCAN;
            while (__hip_atomic_load(counter, __ATOMIC_ACQUIRE,
                                     __HIP_MEMORY_SCOPE_AGENT) < gen) {
                __builtin_amdgcn_s_sleep(2);
            }
        }
        __syncthreads();
    }

    // final cell state
#pragma unroll
    for (int j = 0; j < 4; ++j) {
        int e = tid * 4 + j;
        int b = e >> 4;
        int n = e & 15;
        cn[b * H_ + nTile + n] = creg[j];
    }
}

// --------------------------- host launcher ----------------------------------

extern "C" void kernel_launch(void* const* d_in, const int* in_sizes, int n_in,
                              void* d_out, int out_size, void* d_ws, size_t ws_size,
                              hipStream_t stream) {
    (void)in_sizes; (void)n_in; (void)out_size; (void)ws_size;

    const float* x    = (const float*)d_in[0];   // [B, T, D]
    const float* Wih0 = (const float*)d_in[1];   // [4H, D]
    const float* b0   = (const float*)d_in[2];   // [4H]
    const float* Whh0 = (const float*)d_in[3];   // [4H, H]
    const float* Wih1 = (const float*)d_in[4];   // [4H, H]
    const float* b1   = (const float*)d_in[5];   // [4H]
    const float* Whh1 = (const float*)d_in[6];   // [4H, H]
    float* out = (float*)d_out;

    // output regions: output [B,T,H], h_n [2,B,H], c_n [2,B,H]
    float* hn0 = out + (size_t)B_ * T_ * H_;
    float* hn1 = hn0 + (size_t)B_ * H_;
    float* cn0 = hn1 + (size_t)B_ * H_;
    float* cn1 = cn0 + (size_t)B_ * H_;

    // workspace layout (all chunks are multiples of 256 bytes)
    char* ws = (char*)d_ws;
    size_t off = 0;
    unsigned* counters = (unsigned*)(ws + off); off += 256;
    bf16_t* x_bf   = (bf16_t*)(ws + off); off += (size_t)T_ * B_ * D_ * 2;   // 32 MB
    bf16_t* wih0_b = (bf16_t*)(ws + off); off += (size_t)G4H * D_ * 2;       //  8 MB
    bf16_t* whh0_b = (bf16_t*)(ws + off); off += (size_t)G4H * H_ * 2;
    bf16_t* wih1_b = (bf16_t*)(ws + off); off += (size_t)G4H * H_ * 2;
    bf16_t* whh1_b = (bf16_t*)(ws + off); off += (size_t)G4H * H_ * 2;
    bf16_t* hs0    = (bf16_t*)(ws + off); off += (size_t)T_ * B_ * H_ * 2;   // 32 MB
    bf16_t* hs1    = (bf16_t*)(ws + off); off += (size_t)T_ * B_ * H_ * 2;
    float*  gxbuf  = (float*)(ws + off);  off += (size_t)T_ * B_ * G4H * 4;  // 256 MB (reused)

    hipLaunchKernelGGL(init_ws_kernel, dim3(1), dim3(32), 0, stream, counters);

    const int nX = B_ * T_ * D_;        // 16,777,216
    const int nW = G4H * D_;            //  4,194,304
    hipLaunchKernelGGL(cvt_f32_to_bf16_kernel, dim3(nX / 256), dim3(256), 0, stream, x,    x_bf,   nX);
    hipLaunchKernelGGL(cvt_f32_to_bf16_kernel, dim3(nW / 256), dim3(256), 0, stream, Wih0, wih0_b, nW);
    hipLaunchKernelGGL(cvt_f32_to_bf16_kernel, dim3(nW / 256), dim3(256), 0, stream, Whh0, whh0_b, nW);
    hipLaunchKernelGGL(cvt_f32_to_bf16_kernel, dim3(nW / 256), dim3(256), 0, stream, Wih1, wih1_b, nW);
    hipLaunchKernelGGL(cvt_f32_to_bf16_kernel, dim3(nW / 256), dim3(256), 0, stream, Whh1, whh1_b, nW);

    dim3 gGemm((T_ * B_) / 64, G4H / 64);   // 256 x 64 blocks

    // layer 0: A = x_bf (batch-first: srcRow = t*1 + b*512)
    hipLaunchKernelGGL(gemm_gx_kernel, gGemm, dim3(128), 0, stream,
                       x_bf, wih0_b, b0, gxbuf, /*strideT=*/1, /*strideB=*/T_);
    hipLaunchKernelGGL(lstm_scan_kernel, dim3(NWG_SCAN), dim3(128), 0, stream,
                       gxbuf, whh0_b, hs0, (float*)nullptr, hn0, cn0,
                       counters + 0, /*writeOut=*/0);

    // layer 1: A = hs0 (already [T,B,H]: srcRow = t*32 + b)
    hipLaunchKernelGGL(gemm_gx_kernel, gGemm, dim3(128), 0, stream,
                       hs0, wih1_b, b1, gxbuf, /*strideT=*/B_, /*strideB=*/1);
    hipLaunchKernelGGL(lstm_scan_kernel, dim3(NWG_SCAN), dim3(128), 0, stream,
                       gxbuf, whh1_b, hs1, out, hn1, cn1,
                       counters + 1, /*writeOut=*/1);
}
